// GlobalAttention_72834055405923
// MI455X (gfx1250) — matrix operand order repaired and checked by reading, not compiled, not run on hardware
//
#include <hip/hip_runtime.h>

// ---------------- problem constants ----------------
#define IDIM   1024
#define ODIM   1024
#define NHEADS 16
#define DH     64          // ODIM / NHEADS
#define BB     4
#define SS     1024
#define NROW   (BB*SS)     // 4096 rows of m_feats
#define NCOL   (3*ODIM)    // 3072 fused projection columns [K | Q | V]

// ---------------- WMMA types ----------------
typedef __attribute__((ext_vector_type(16))) __bf16 v16bf;
typedef __attribute__((ext_vector_type(8)))  float  v8f;

struct BF16Frag { union { unsigned u[8]; v16bf v; }; };

__device__ inline v8f wmma_bf16(v16bf a, v16bf b, v8f c) {
  // v_wmma_f32_16x16x32_bf16: D = A(16x32) * B(32x16) + C, f32 accumulate
  return __builtin_amdgcn_wmma_f32_16x16x32_bf16(false, a, false, b, (short)0, c, false, false);
}

__device__ inline unsigned short f2bf(float f) {
  unsigned u = __float_as_uint(f);
  u += 0x7fffu + ((u >> 16) & 1u);   // round-to-nearest-even
  return (unsigned short)(u >> 16);
}
__device__ inline unsigned pack2bf(float lo, float hi) {
  return (unsigned)f2bf(lo) | ((unsigned)f2bf(hi) << 16);
}

// A-matrix 16x32 bf16 fragment from row-major bf16 memory.
// lane: m = lane&15 row; half = lane>>4.
// e in [0,8)  -> K = kbase + 8*half + e        (u[0..3])
// e in [8,16) -> K = kbase + 16 + 8*half + e-8 (u[4..7])
__device__ inline v16bf load_frag_a_bf16(const unsigned short* rowPtr, int kbase, int half) {
  uint4 a = *(const uint4*)(rowPtr + kbase + half * 8);
  uint4 b = *(const uint4*)(rowPtr + kbase + 16 + half * 8);
  BF16Frag f;
  f.u[0]=a.x; f.u[1]=a.y; f.u[2]=a.z; f.u[3]=a.w;
  f.u[4]=b.x; f.u[5]=b.y; f.u[6]=b.z; f.u[7]=b.w;
  return f.v;
}

// B-matrix 32x16 bf16 fragment; rowPtr points at the K-major row for column n=lane&15.
// e in [0,8) -> K = kbase + 8*half + e; e in [8,16) -> K = kbase + 16 + 8*half + e-8
__device__ inline v16bf load_frag_b_bf16(const unsigned short* rowPtr, int kbase, int half) {
  uint4 a = *(const uint4*)(rowPtr + kbase + half * 8);
  uint4 b = *(const uint4*)(rowPtr + kbase + 16 + half * 8);
  BF16Frag f;
  f.u[0]=a.x; f.u[1]=a.y; f.u[2]=a.z; f.u[3]=a.w;
  f.u[4]=b.x; f.u[5]=b.y; f.u[6]=b.z; f.u[7]=b.w;
  return f.v;
}

// A fragment built from f32 data in LDS (softmax weights), packed to bf16 in-register.
__device__ inline v16bf load_frag_a_f32(const float* rowPtr, int kbase, int half) {
  const float4* p0 = (const float4*)(rowPtr + kbase + half * 8);
  const float4* p1 = (const float4*)(rowPtr + kbase + 16 + half * 8);
  float4 a0 = p0[0], a1 = p0[1];
  float4 b0 = p1[0], b1 = p1[1];
  BF16Frag f;
  f.u[0]=pack2bf(a0.x,a0.y); f.u[1]=pack2bf(a0.z,a0.w);
  f.u[2]=pack2bf(a1.x,a1.y); f.u[3]=pack2bf(a1.z,a1.w);
  f.u[4]=pack2bf(b0.x,b0.y); f.u[5]=pack2bf(b0.z,b0.w);
  f.u[6]=pack2bf(b1.x,b1.y); f.u[7]=pack2bf(b1.z,b1.w);
  return f.v;
}

// ================= kernel 0: f32 -> bf16 conversion, weight transpose =================
__global__ __launch_bounds__(256) void k_convert(const float* __restrict__ mf,
                                                 const float* __restrict__ Wc,
                                                 const float* __restrict__ Wv,
                                                 unsigned short* __restrict__ Xb,
                                                 unsigned short* __restrict__ WbT) {
  int idx = blockIdx.x * 256 + threadIdx.x;
  const int NX = NROW * IDIM;            // 4194304
  const int NW = NCOL * IDIM;            // 3145728
  if (idx < NX) {
    Xb[idx] = f2bf(mf[idx]);
  } else if (idx < NX + NW) {
    int t = idx - NX;
    int n = t >> 10, k = t & 1023;       // WbT[n][k]
    float v = (n < ODIM) ? Wv[(size_t)k * ODIM + n]
                         : Wc[(size_t)k * (2 * ODIM) + (n - ODIM)];
    WbT[t] = f2bf(v);
  }
}

// ================= kernel 1: fused projection GEMM (ping-pong pipelined) =================
// C(4096 x 3072) = Xb * WbT^T + bias; scatter to Kb (b,i,h,d), Qb (b,j,h,d), Vt (b,h*64+d,s)
__global__ __launch_bounds__(256) void k_proj(const unsigned short* __restrict__ Xb,
                                              const unsigned short* __restrict__ WbT,
                                              const float* __restrict__ b_c,
                                              const float* __restrict__ b_v,
                                              unsigned short* __restrict__ Kb,
                                              unsigned short* __restrict__ Qb,
                                              unsigned short* __restrict__ Vt) {
  __shared__ float tile[8 * 256];                       // 16x16 f32 stage per wave
  const int wave = threadIdx.x >> 5, lane = threadIdx.x & 31;
  const int half = lane >> 4, n = lane & 15;
  const int task = blockIdx.x * 8 + wave;               // 12288 wave-tasks
  const int ngroup = task % (NCOL / 64);                // 48 column groups of 64
  const int mtile  = task / (NCOL / 64);                // 256 row tiles of 16
  const int i0 = mtile * 16;

  const unsigned short* arow = Xb + (size_t)(i0 + (lane & 15)) * IDIM;
  const unsigned short* brow[4];
#pragma unroll
  for (int t = 0; t < 4; ++t)
    brow[t] = WbT + (size_t)(ngroup * 64 + t * 16 + n) * IDIM;

  v8f acc[4] = {};
  // ping-pong double buffers: each buffer written only by loads (no reg copies)
  v16bf a0 = load_frag_a_bf16(arow, 0, half);
  v16bf p0[4];
#pragma unroll
  for (int t = 0; t < 4; ++t) p0[t] = load_frag_b_bf16(brow[t], 0, half);

  for (int k = 0; k < IDIM; k += 64) {                  // 16 iterations, 2 K-steps each
    // prefetch k+32 into buffer 1 (always in range: k <= 960)
    v16bf a1 = load_frag_a_bf16(arow, k + 32, half);
    v16bf p1[4];
#pragma unroll
    for (int t = 0; t < 4; ++t) p1[t] = load_frag_b_bf16(brow[t], k + 32, half);
#pragma unroll
    for (int t = 0; t < 4; ++t) acc[t] = wmma_bf16(a0, p0[t], acc[t]);

    // prefetch k+64 into buffer 0 (clamped on last iteration)
    int k2 = (k + 64 < IDIM) ? k + 64 : 0;
    a0 = load_frag_a_bf16(arow, k2, half);
#pragma unroll
    for (int t = 0; t < 4; ++t) p0[t] = load_frag_b_bf16(brow[t], k2, half);
#pragma unroll
    for (int t = 0; t < 4; ++t) acc[t] = wmma_bf16(a1, p1[t], acc[t]);
  }

  float* my = tile + wave * 256;
  for (int t = 0; t < 4; ++t) {
    __syncthreads();
#pragma unroll
    for (int r = 0; r < 8; ++r)
      my[(r + 8 * half) * 16 + n] = acc[t][r];
    __syncthreads();

    const int col0 = ngroup * 64 + t * 16;
    if (col0 < 2 * ODIM) {
      // K (cols 0..1023, bias b_v) or Q (cols 1024..2047, bias b_c[0..1023])
      const float* bias = (col0 < ODIM) ? (b_v + col0) : (b_c + (col0 - ODIM));
      unsigned short* dst = (col0 < ODIM) ? (Kb + col0) : (Qb + (col0 - ODIM));
      int rr = lane >> 1, cb = (lane & 1) * 8;
      union { unsigned short s[8]; uint4 q; } pk;
#pragma unroll
      for (int e = 0; e < 8; ++e)
        pk.s[e] = f2bf(my[rr * 16 + cb + e] + bias[cb + e]);
      *(uint4*)(dst + (size_t)(i0 + rr) * ODIM + cb) = pk.q;
    } else {
      // V: store transposed, Vt[(b*1024 + hd)][s], bias b_c[1024 + hd]
      int vbase = col0 - 2 * ODIM;
      int cc = lane >> 1, sh = (lane & 1) * 8;
      int b = i0 >> 10, s0 = i0 & 1023;
      float bias = b_c[ODIM + vbase + cc];
      union { unsigned short s[8]; uint4 q; } pk;
#pragma unroll
      for (int e = 0; e < 8; ++e)
        pk.s[e] = f2bf(my[(sh + e) * 16 + cc] + bias);
      *(uint4*)(Vt + (size_t)(b * ODIM + vbase + cc) * SS + s0 + sh) = pk.q;
    }
  }
}

// ================= kernel 2: fused attention =================
// grid (64, 16), block 256 (8 waves = 4 strip-pairs). LDS: 4x(16x1024) f32 scores + mask.
// Two waves cooperate per 16-row strip: split j-range (scores), rows (softmax),
// d-tiles (PV GEMM). 260 KB LDS -> 1 block/WGP -> 2 waves per SIMD32.
__global__ __launch_bounds__(256) void k_attn(const unsigned short* __restrict__ Kb,
                                              const unsigned short* __restrict__ Qb,
                                              const unsigned short* __restrict__ Vt,
                                              const int* __restrict__ mask,
                                              const float* __restrict__ m_feats,
                                              float* __restrict__ out_updated,
                                              float* __restrict__ out_w) {
  extern __shared__ float smem[];                 // [4*16384 scores][1024 maskflag]
  float* mkf = smem + 4 * 16384;
  const int wave = threadIdx.x >> 5, lane = threadIdx.x & 31;
  const int strip = wave >> 1, sub = wave & 1;    // 4 strips, 2 waves each
  const int half = lane >> 4, n = lane & 15;
  const int b = blockIdx.x >> 4, h = blockIdx.x & 15;
  const int itile = blockIdx.y * 4 + strip;
  const int i0 = itile * 16;

  for (int j = threadIdx.x; j < SS; j += 256)
    mkf[j] = (mask[b * SS + j] == 0) ? 1.f : 0.f;
  __syncthreads();

  // ---- phase 1: scores = K_h * Q_h^T * (1/sqrt(dh)) into LDS; wave covers half the j range
  const unsigned short* arow = Kb + (size_t)(b * SS + i0 + (lane & 15)) * ODIM + h * DH;
  v16bf aK0 = load_frag_a_bf16(arow, 0, half);
  v16bf aK1 = load_frag_a_bf16(arow, 32, half);
  float* srow = smem + strip * 16384;
  const unsigned short* qbase = Qb + (size_t)(b * SS) * ODIM + h * DH;
  const int jt0 = sub * 32, jend = jt0 + 32;

  v16bf qa0 = load_frag_b_bf16(qbase + (size_t)(jt0 * 16 + n) * ODIM, 0, half);
  v16bf qa1 = load_frag_b_bf16(qbase + (size_t)(jt0 * 16 + n) * ODIM, 32, half);
  for (int jt = jt0; jt < jend; jt += 2) {        // 16 iterations, ping-pong
    // prefetch jt+1 into buffer B (always in range: range length is even)
    const unsigned short* nb1 = qbase + (size_t)((jt + 1) * 16 + n) * ODIM;
    v16bf qb0 = load_frag_b_bf16(nb1, 0, half);
    v16bf qb1 = load_frag_b_bf16(nb1, 32, half);
    v8f c = {};
    c = wmma_bf16(aK0, qa0, c);
    c = wmma_bf16(aK1, qa1, c);
#pragma unroll
    for (int r = 0; r < 8; ++r)
      srow[(r + 8 * half) * SS + jt * 16 + n] = c[r] * 0.125f;     // 1/sqrt(64)

    // prefetch jt+2 into buffer A (clamped on last iteration)
    int j2 = (jt + 2 < jend) ? jt + 2 : jt0;
    const unsigned short* nb2 = qbase + (size_t)(j2 * 16 + n) * ODIM;
    qa0 = load_frag_b_bf16(nb2, 0, half);
    qa1 = load_frag_b_bf16(nb2, 32, half);
    v8f c2 = {};
    c2 = wmma_bf16(aK0, qb0, c2);
    c2 = wmma_bf16(aK1, qb1, c2);
#pragma unroll
    for (int r = 0; r < 8; ++r)
      srow[(r + 8 * half) * SS + (jt + 1) * 16 + n] = c2[r] * 0.125f;
  }
  __syncthreads();

  // ---- phase 2: masked softmax over j, in LDS; each wave does 8 of the strip's 16 rows
  float* wout_base = out_w + ((size_t)(b * NHEADS + h) * SS + i0) * SS;
  for (int mm = 0; mm < 8; ++mm) {
    int m = sub * 8 + mm;
    float* row = srow + m * SS;
    float vmax = -3.0e38f;
    for (int e = 0; e < 32; ++e) {
      int j = e * 32 + lane;
      float s = row[j];
      s = (mkf[j] != 0.f) ? -1.0e9f : s;
      row[j] = s;
      vmax = fmaxf(vmax, s);
    }
#pragma unroll
    for (int off = 16; off > 0; off >>= 1)
      vmax = fmaxf(vmax, __shfl_xor(vmax, off, 32));
    float sum = 0.f;
    for (int e = 0; e < 32; ++e) {
      int j = e * 32 + lane;
      float p = __expf(row[j] - vmax);
      row[j] = p;
      sum += p;
    }
#pragma unroll
    for (int off = 16; off > 0; off >>= 1)
      sum += __shfl_xor(sum, off, 32);
    float inv = 1.f / sum;
    float* wo = wout_base + (size_t)m * SS;
    for (int e = 0; e < 32; ++e) {
      int j = e * 32 + lane;
      float wv = row[j] * inv;
      row[j] = wv;
      wo[j] = wv;                                 // coalesced 128B stores
    }
  }
  __syncthreads();

  // ---- phase 3: r = w_strip (LDS) @ V_h; wave computes 2 of the strip's 4 d-tiles
  v8f acc[2] = {};
  const float* arow2 = srow + (lane & 15) * SS;
  const unsigned short* vrow0 = Vt + (size_t)(b * ODIM + h * DH + (sub * 2 + 0) * 16 + n) * SS;
  const unsigned short* vrow1 = Vt + (size_t)(b * ODIM + h * DH + (sub * 2 + 1) * 16 + n) * SS;
  v16bf va0 = load_frag_b_bf16(vrow0, 0, half);
  v16bf va1 = load_frag_b_bf16(vrow1, 0, half);
  for (int k = 0; k < SS; k += 64) {              // 16 iterations, ping-pong
    // prefetch k+32 into buffer B (always in range: k <= 960)
    v16bf vb0 = load_frag_b_bf16(vrow0, k + 32, half);
    v16bf vb1 = load_frag_b_bf16(vrow1, k + 32, half);
    v16bf aA = load_frag_a_f32(arow2, k, half);
    acc[0] = wmma_bf16(aA, va0, acc[0]);
    acc[1] = wmma_bf16(aA, va1, acc[1]);

    // prefetch k+64 into buffer A (clamped on last iteration)
    int k2 = (k + 64 < SS) ? k + 64 : 0;
    va0 = load_frag_b_bf16(vrow0, k2, half);
    va1 = load_frag_b_bf16(vrow1, k2, half);
    v16bf aB = load_frag_a_f32(arow2, k + 32, half);
    acc[0] = wmma_bf16(aB, vb0, acc[0]);
    acc[1] = wmma_bf16(aB, vb1, acc[1]);
  }
  __syncthreads();

  // ---- epilogue: residual add + store (per-wave 256-float stage in now-free LDS)
  float* tilebuf = smem + wave * 256;
  for (int tt = 0; tt < 2; ++tt) {
    int t = sub * 2 + tt;
    __syncthreads();
#pragma unroll
    for (int r = 0; r < 8; ++r)
      tilebuf[(r + 8 * half) * 16 + n] = acc[tt][r];
    __syncthreads();
    int rr = lane >> 1, cb = (lane & 1) * 8;
    size_t oidx = (size_t)(b * SS + i0 + rr) * ODIM + h * DH + t * 16 + cb;
    float4 r0 = *(const float4*)(m_feats + oidx);
    float4 r1 = *(const float4*)(m_feats + oidx + 4);
    float4 o0, o1;
    o0.x = tilebuf[rr*16+cb+0] + r0.x; o0.y = tilebuf[rr*16+cb+1] + r0.y;
    o0.z = tilebuf[rr*16+cb+2] + r0.z; o0.w = tilebuf[rr*16+cb+3] + r0.w;
    o1.x = tilebuf[rr*16+cb+4] + r1.x; o1.y = tilebuf[rr*16+cb+5] + r1.y;
    o1.z = tilebuf[rr*16+cb+6] + r1.z; o1.w = tilebuf[rr*16+cb+7] + r1.w;
    *(float4*)(out_updated + oidx)     = o0;
    *(float4*)(out_updated + oidx + 4) = o1;
  }
}

// ================= host launch =================
extern "C" void kernel_launch(void* const* d_in, const int* in_sizes, int n_in,
                              void* d_out, int out_size, void* d_ws, size_t ws_size,
                              hipStream_t stream) {
  (void)in_sizes; (void)n_in; (void)out_size; (void)ws_size;
  const float* m_feats = (const float*)d_in[0];
  const int*   mask    = (const int*)d_in[1];
  const float* W_c     = (const float*)d_in[2];
  const float* b_c     = (const float*)d_in[3];
  const float* W_v     = (const float*)d_in[4];
  const float* b_v     = (const float*)d_in[5];

  float* out_updated = (float*)d_out;                          // (4,1024,1024) f32
  float* out_w = out_updated + (size_t)BB * SS * ODIM;         // (4,16,1024,1024) f32

  // workspace layout (bf16 as ushort): 38 MB total
  unsigned short* Xb  = (unsigned short*)d_ws;                 // 4096 x 1024
  unsigned short* WbT = Xb  + (size_t)NROW * IDIM;             // 3072 x 1024 (transposed)
  unsigned short* Kb  = WbT + (size_t)NCOL * IDIM;             // 4096 x 1024 (b,i,h,d)
  unsigned short* Qb  = Kb  + (size_t)NROW * ODIM;             // 4096 x 1024 (b,j,h,d)
  unsigned short* Vt  = Qb  + (size_t)NROW * ODIM;             // (b,h*64+d) x 1024 (s)

  // k0: conversion
  {
    int total = NROW * IDIM + NCOL * IDIM;
    int blocks = (total + 255) / 256;
    k_convert<<<blocks, 256, 0, stream>>>(m_feats, W_c, W_v, Xb, WbT);
  }
  // k1: projection GEMM (12288 wave-tasks, 8 waves/block)
  k_proj<<<(NROW / 16) * (NCOL / 64) / 8, 256, 0, stream>>>(Xb, WbT, b_c, b_v, Kb, Qb, Vt);

  // k2: fused attention; 260 KB dynamic LDS per workgroup (<= 320 KB WGP limit)
  size_t shbytes = (size_t)(4 * 16384 + 1024) * sizeof(float); // 266240
  hipFuncSetAttribute((const void*)k_attn, hipFuncAttributeMaxDynamicSharedMemorySize,
                      (int)shbytes);
  k_attn<<<dim3(BB * NHEADS, SS / 16 / 4), 256, shbytes, stream>>>(
      Kb, Qb, Vt, mask, m_feats, out_updated, out_w);
}
